// JMAC_41154376630473
// MI455X (gfx1250) — matrix operand order repaired and checked by reading, not compiled, not run on hardware
//
#include <hip/hip_runtime.h>
#include <hip/hip_bf16.h>
#include <math.h>

#define NN 50000
#define EE 400000
#define RRR 500
#define DD 128
#define NAMED 300
#define NAMEDP 320   // padded to multiple of 32
#define SLOPE 0.2f
#define BN_EPS 1e-5f

typedef __bf16 bf16_t;
typedef bf16_t v16bf __attribute__((ext_vector_type(16)));
typedef bf16_t v2bf __attribute__((ext_vector_type(2)));
typedef float v8f __attribute__((ext_vector_type(8)));

union BF16x16 { v16bf v; unsigned short s[16]; uint4 q[2]; };

__device__ __forceinline__ unsigned short f2bf(float f) {
  unsigned u = __float_as_uint(f);
  unsigned r = u + 0x7fffu + ((u >> 16) & 1u);   // round-to-nearest-even
  return (unsigned short)(r >> 16);
}

// packed f32x2 -> bf16x2 conversion (hardware v_cvt_pk_bf16_f32 when available)
#if __has_builtin(__builtin_amdgcn_cvt_pk_bf16_f32)
__device__ __forceinline__ void cvt2(unsigned short* dst, float x, float y) {
  union { v2bf v; unsigned short s[2]; } u;
  u.v = __builtin_amdgcn_cvt_pk_bf16_f32(x, y);
  dst[0] = u.s[0]; dst[1] = u.s[1];
}
#else
__device__ __forceinline__ void cvt2(unsigned short* dst, float x, float y) {
  dst[0] = f2bf(x); dst[1] = f2bf(y);
}
#endif

// order-preserving float <-> uint encoding for atomicMax
__device__ __forceinline__ unsigned encf(float f) {
  unsigned u = __float_as_uint(f);
  return (u & 0x80000000u) ? ~u : (u | 0x80000000u);
}
__device__ __forceinline__ float decf(unsigned u) {
  return (u & 0x80000000u) ? __uint_as_float(u & 0x7fffffffu) : __uint_as_float(~u);
}

// ---------------------------------------------------------------------------
// B prepack: fp32 [Kp,128] (rows >= Korig treated as 0) -> bf16 fragments in
// WMMA B order: Bp[((kc*8+nt)*32+lane)*16 + i] = B[(kc*32+(lane>>4)*8+(i&7)+(i>>3)*16)*128
//                                                  + nt*16 + (lane&15)]
// ---------------------------------------------------------------------------
__global__ void prepack_b(const float* __restrict__ B, unsigned short* __restrict__ Bp,
                          int Kp, int Korig) {
  int idx = blockIdx.x * blockDim.x + threadIdx.x;
  if (idx >= Kp * DD) return;
  int i    = idx & 15;
  int lane = (idx >> 4) & 31;
  int ntkc = idx >> 9;            // kc*8 + nt
  int nt = ntkc & 7, kc = ntkc >> 3;
  int k = kc * 32 + (lane >> 4) * 8 + (i & 7) + ((i >> 3) << 4);
  int n = nt * 16 + (lane & 15);
  Bp[idx] = (k < Korig) ? f2bf(B[(size_t)k * DD + n]) : (unsigned short)0;
}

// ---------------------------------------------------------------------------
// C[M,128] = act(A[M,K] @ B[K,128]) (+= if ACC). One wave32 per 16-row band;
// branchless inner loop; B consumed in groups of 4 tiles for load/WMMA overlap.
// ---------------------------------------------------------------------------
template <int KK, bool ACC, bool ACT>
__global__ __launch_bounds__(32)
void gemm_nk128_t(const float* __restrict__ A, int M,
                  const unsigned short* __restrict__ Bp,
                  float* __restrict__ C)
{
  const int lane = threadIdx.x & 31;
  const int half = lane >> 4;
  const int l16  = lane & 15;
  const int row0 = blockIdx.x << 4;
  const bool fullTile = (row0 + 16 <= M);

  v8f acc[8];
  if (ACC) {
    if (fullTile) {
#pragma unroll
      for (int nt = 0; nt < 8; ++nt)
#pragma unroll
        for (int v = 0; v < 8; ++v)
          acc[nt][v] = C[(size_t)(row0 + v + 8 * half) * DD + nt * 16 + l16];
    } else {
#pragma unroll
      for (int nt = 0; nt < 8; ++nt)
#pragma unroll
        for (int v = 0; v < 8; ++v) {
          int r = row0 + v + 8 * half;
          if (r >= M) r = M - 1;
          acc[nt][v] = C[(size_t)r * DD + nt * 16 + l16];
        }
    }
  } else {
#pragma unroll
    for (int nt = 0; nt < 8; ++nt)
#pragma unroll
      for (int v = 0; v < 8; ++v) acc[nt][v] = 0.f;
  }

  int arow = row0 + l16; if (arow >= M) arow = M - 1;
  const float* __restrict__ Arow = A + (size_t)arow * KK;

  constexpr int nkc = KK >> 5;
  for (int kc = 0; kc < nkc; ++kc) {
    const int k0 = kc << 5;
    // A fragment 16x32 bf16: lanes 0-15 row M=lane:    e0..7=K0..7,  e8..15=K16..23
    //                        lanes 16-31 row M=lane-16: e0..7=K8..15, e8..15=K24..31
    const float4 f0 = *(const float4*)(Arow + k0 + half * 8);
    const float4 f1 = *(const float4*)(Arow + k0 + half * 8 + 4);
    const float4 f2 = *(const float4*)(Arow + k0 + 16 + half * 8);
    const float4 f3 = *(const float4*)(Arow + k0 + 16 + half * 8 + 4);
    BF16x16 a;
    cvt2(&a.s[0],  f0.x, f0.y); cvt2(&a.s[2],  f0.z, f0.w);
    cvt2(&a.s[4],  f1.x, f1.y); cvt2(&a.s[6],  f1.z, f1.w);
    cvt2(&a.s[8],  f2.x, f2.y); cvt2(&a.s[10], f2.z, f2.w);
    cvt2(&a.s[12], f3.x, f3.y); cvt2(&a.s[14], f3.z, f3.w);
#pragma unroll
    for (int g = 0; g < 2; ++g) {
      BF16x16 b[4];
#pragma unroll
      for (int j = 0; j < 4; ++j) {
        const uint4* q =
            (const uint4*)(Bp + ((size_t)((kc * 8 + g * 4 + j) * 32 + lane) << 4));
        b[j].q[0] = q[0];
        b[j].q[1] = q[1];
      }
#pragma unroll
      for (int j = 0; j < 4; ++j)
        acc[g * 4 + j] = __builtin_amdgcn_wmma_f32_16x16x32_bf16(
            false, a.v, false, b[j].v, (short)0, acc[g * 4 + j], false, false);
    }
  }

  if (ACT) {
#pragma unroll
    for (int nt = 0; nt < 8; ++nt)
#pragma unroll
      for (int v = 0; v < 8; ++v) {
        float val = acc[nt][v];
        acc[nt][v] = (val >= 0.f) ? val : val * SLOPE;
      }
  }

  if (fullTile) {
#pragma unroll
    for (int nt = 0; nt < 8; ++nt)
#pragma unroll
      for (int v = 0; v < 8; ++v)
        C[(size_t)(row0 + v + 8 * half) * DD + nt * 16 + l16] = acc[nt][v];
  } else {
#pragma unroll
    for (int nt = 0; nt < 8; ++nt)
#pragma unroll
      for (int v = 0; v < 8; ++v) {
        int r = row0 + v + 8 * half;
        if (r < M) C[(size_t)r * DD + nt * 16 + l16] = acc[nt][v];
      }
  }
}

// ---------------------------------------------------------------------------
// Elementwise / edge kernels
// ---------------------------------------------------------------------------
__global__ void fill_u32(unsigned* __restrict__ p, unsigned v, size_t n) {
  size_t i = (size_t)blockIdx.x * blockDim.x + threadIdx.x;
  if (i < n) p[i] = v;
}

__global__ void pad_copy_kernel(const float* __restrict__ src, float* __restrict__ dst,
                                int n, int ksrc, int kdst) {
  size_t i = (size_t)blockIdx.x * blockDim.x + threadIdx.x;
  if (i >= (size_t)n * kdst) return;
  int c = (int)(i % kdst);
  size_t r = i / kdst;
  dst[i] = (c < ksrc) ? src[r * ksrc + c] : 0.f;
}

__global__ void relcat_kernel(const float* __restrict__ rel, const float* __restrict__ loop,
                              int Rr, float* __restrict__ out) {
  int i = blockIdx.x * blockDim.x + threadIdx.x;
  if (i >= (Rr + 1) * DD) return;
  int r = i >> 7, d = i & 127;
  out[i] = (r < Rr) ? rel[i] : loop[d];
}

__global__ __launch_bounds__(256)
void normalize_rows(const float* __restrict__ x, float* __restrict__ out, int n) {
  int row  = (blockIdx.x * blockDim.x + threadIdx.x) >> 5;
  int lane = threadIdx.x & 31;
  if (row >= n) return;
  const float4 v = *(const float4*)(x + (size_t)row * DD + lane * 4);
  float p = v.x * v.x + v.y * v.y + v.z * v.z + v.w * v.w;
#pragma unroll
  for (int off = 16; off; off >>= 1) p += __shfl_down(p, off, 32);
  p = __shfl(p, 0, 32);
  float inv = 1.f / fmaxf(sqrtf(p), 1e-12f);
  float4 o = { v.x * inv, v.y * inv, v.z * inv, v.w * inv };
  *(float4*)(out + (size_t)row * DD + lane * 4) = o;
}

__global__ void deg_count(const int* __restrict__ eidx, int e_cnt, float* __restrict__ deg) {
  int e = blockIdx.x * blockDim.x + threadIdx.x;
  if (e >= e_cnt) return;
  unsafeAtomicAdd(deg + eidx[e], 1.0f);
}

__global__ void deg_inv_kernel(const float* __restrict__ deg, float* __restrict__ dinv, int n) {
  int i = blockIdx.x * blockDim.x + threadIdx.x;
  if (i >= n) return;
  float d = deg[i];
  dinv[i] = (d > 0.f) ? rsqrtf(fmaxf(d, 1.f)) : 0.f;
}

// score_e = leaky(y1[dst] + y2[src] - relW[t]) . aatt ; segment-max via atomicMax
__global__ __launch_bounds__(256)
void edge_score_kernel(const int* __restrict__ eidx, const int* __restrict__ etype, int e_cnt,
                       const float* __restrict__ y1, const float* __restrict__ y2,
                       const float* __restrict__ relw, const float* __restrict__ aatt,
                       float* __restrict__ score, unsigned* __restrict__ smax)
{
  int e    = (blockIdx.x * blockDim.x + threadIdx.x) >> 5;
  int lane = threadIdx.x & 31;
  if (e >= e_cnt) return;
  int d = eidx[e], s = eidx[e_cnt + e], t = etype[e];
  const float4 v1 = *(const float4*)(y1 + (size_t)d * DD + lane * 4);
  const float4 v2 = *(const float4*)(y2 + (size_t)s * DD + lane * 4);
  const float4 vr = *(const float4*)(relw + (size_t)t * DD + lane * 4);
  const float4 va = *(const float4*)(aatt + lane * 4);
  float p = 0.f, u;
  u = v1.x + v2.x - vr.x; u = (u >= 0.f) ? u : u * SLOPE; p += u * va.x;
  u = v1.y + v2.y - vr.y; u = (u >= 0.f) ? u : u * SLOPE; p += u * va.y;
  u = v1.z + v2.z - vr.z; u = (u >= 0.f) ? u : u * SLOPE; p += u * va.z;
  u = v1.w + v2.w - vr.w; u = (u >= 0.f) ? u : u * SLOPE; p += u * va.w;
#pragma unroll
  for (int off = 16; off; off >>= 1) p += __shfl_down(p, off, 32);
  if (lane == 0) {
    score[e] = p;
    atomicMax(smax + d, encf(p));
  }
}

__global__ void edge_exp_kernel(const int* __restrict__ eidx, int e_cnt,
                                const unsigned* __restrict__ smax,
                                float* __restrict__ score, float* __restrict__ denom) {
  int e = blockIdx.x * blockDim.x + threadIdx.x;
  if (e >= e_cnt) return;
  int d = eidx[e];
  float ex = expf(score[e] - decf(smax[d]));
  score[e] = ex;
  unsafeAtomicAdd(denom + d, ex);
}

// neigh[dst] += (z[src] - relG[t]) * (alpha * deg_inv[dst])
__global__ __launch_bounds__(256)
void edge_msg_kernel(const int* __restrict__ eidx, const int* __restrict__ etype, int e_cnt,
                     const float* __restrict__ z, const float* __restrict__ relg,
                     const float* __restrict__ score, const float* __restrict__ denom,
                     const float* __restrict__ deginv, float* __restrict__ neigh)
{
  int e    = (blockIdx.x * blockDim.x + threadIdx.x) >> 5;
  int lane = threadIdx.x & 31;
  if (e >= e_cnt) return;
  int d = eidx[e], s = eidx[e_cnt + e], t = etype[e];
  float coef = score[e] / denom[d] * deginv[d];
  const float4 vz = *(const float4*)(z + (size_t)s * DD + lane * 4);
  const float4 vr = *(const float4*)(relg + (size_t)t * DD + lane * 4);
  float* o = neigh + (size_t)d * DD + lane * 4;
  unsafeAtomicAdd(o + 0, (vz.x - vr.x) * coef);
  unsafeAtomicAdd(o + 1, (vz.y - vr.y) * coef);
  unsafeAtomicAdd(o + 2, (vz.z - vr.z) * coef);
  unsafeAtomicAdd(o + 3, (vz.w - vr.w) * coef);
}

// h = (neigh + z - relG[last]) * 0.5 (in place), plus per-dim BN partial sums
__global__ __launch_bounds__(128)
void node_h_stats(float* __restrict__ h, const float* __restrict__ z,
                  const float* __restrict__ relg_last, int n,
                  float* __restrict__ gsum, float* __restrict__ gsumsq)
{
  int d  = threadIdx.x;
  int r0 = blockIdx.x * 64;
  float rl = relg_last[d];
  float s = 0.f, ss = 0.f;
  int rend = r0 + 64; if (rend > n) rend = n;
  for (int r = r0; r < rend; ++r) {
    size_t idx = (size_t)r * DD + d;
    float v = (h[idx] + z[idx] - rl) * 0.5f;
    h[idx] = v;
    s += v; ss += v * v;
  }
  unsafeAtomicAdd(gsum + d, s);
  unsafeAtomicAdd(gsumsq + d, ss);
}

__global__ void bn_finalize(const float* __restrict__ gsum, const float* __restrict__ gsumsq,
                            const float* __restrict__ gamma, const float* __restrict__ beta,
                            int n, float* __restrict__ scale, float* __restrict__ shift) {
  int d = threadIdx.x;
  float inv_n = 1.f / (float)n;
  float mu  = gsum[d] * inv_n;
  float var = gsumsq[d] * inv_n - mu * mu;
  float sc  = gamma[d] * rsqrtf(var + BN_EPS);
  scale[d] = sc;
  shift[d] = beta[d] - mu * sc;
}

__global__ void bn_apply_tanh(const float* __restrict__ h, const float* __restrict__ scale,
                              const float* __restrict__ shift, float* __restrict__ out, size_t total) {
  size_t i = (size_t)blockIdx.x * blockDim.x + threadIdx.x;
  if (i >= total) return;
  int d = (int)(i & 127);
  out[i] = tanhf(h[i] * scale[d] + shift[d]);
}

// ---------------------------------------------------------------------------
// Host-side orchestration
// ---------------------------------------------------------------------------
struct Ws {
  float *B0, *B1, *B2, *B3;   // persistent node buffers [N,128]
  float *y1, *y2, *z;         // conv scratch [N,128]; neigh aliases y1
  float *info_pad;            // [N,320]
  unsigned short *bpack;      // [320*128] bf16 prepacked B
  float *score;               // [E]
  float *denom, *deg, *deginv;// [N]
  unsigned *smax;             // [N]
  float *relcat, *reltmp, *rel2, *relw, *relg;  // [501,128]
  float *arel_t, *arel;       // [500,128]
  float *gsum, *gsumsq, *bnsc, *bnsh;           // [128]
};

static inline void gemm(hipStream_t s, Ws& W, const float* A, int M, int K, int Korig,
                        const float* B, float* C, int accumulate, int act = 0) {
  prepack_b<<<(K * DD + 255) / 256, 256, 0, s>>>(B, W.bpack, K, Korig);
  dim3 grid((M + 15) / 16);
  if (K == NAMEDP)
    gemm_nk128_t<NAMEDP, false, false><<<grid, 32, 0, s>>>(A, M, W.bpack, C);
  else if (accumulate)
    gemm_nk128_t<DD, true, false><<<grid, 32, 0, s>>>(A, M, W.bpack, C);
  else if (act)
    gemm_nk128_t<DD, false, true><<<grid, 32, 0, s>>>(A, M, W.bpack, C);
  else
    gemm_nk128_t<DD, false, false><<<grid, 32, 0, s>>>(A, M, W.bpack, C);
}

static void run_conv(hipStream_t s, Ws& W, const float* x, const float* rel_in,
                     const float* w1, const float* w2, const float* gcnw,
                     const float* loop, const float* watt, const float* aatt,
                     const float* gamma, const float* beta,
                     const int* eidx, const int* etype, float* out)
{
  const int R1 = RRR + 1;
  // relation transform: rel' = leaky(cat(rel, loop) @ w1) @ w2
  relcat_kernel<<<(R1 * DD + 255) / 256, 256, 0, s>>>(rel_in, loop, RRR, W.relcat);
  gemm(s, W, W.relcat, R1, DD, DD, w1, W.reltmp, 0, 1);
  gemm(s, W, W.reltmp, R1, DD, DD, w2, W.rel2, 0, 0);
  // relation projections: relW = rel' @ watt_bot ; relG = rel' @ gcnW
  gemm(s, W, W.rel2, R1, DD, DD, watt + DD * DD, W.relw, 0, 0);
  gemm(s, W, W.rel2, R1, DD, DD, gcnw, W.relg, 0, 0);
  // node projections: y1 = x @ watt_top ; y2 = x @ watt_bot ; z = x @ gcnW
  gemm(s, W, x, NN, DD, DD, watt, W.y1, 0, 0);
  gemm(s, W, x, NN, DD, DD, watt + DD * DD, W.y2, 0, 0);
  gemm(s, W, x, NN, DD, DD, gcnw, W.z, 0, 0);
  // attention: segment softmax over dst
  fill_u32<<<(NN + 255) / 256, 256, 0, s>>>(W.smax, 0u, NN);
  fill_u32<<<(NN + 255) / 256, 256, 0, s>>>((unsigned*)W.denom, 0u, NN);
  edge_score_kernel<<<(EE + 7) / 8, 256, 0, s>>>(eidx, etype, EE, W.y1, W.y2, W.relw,
                                                 aatt, W.score, W.smax);
  edge_exp_kernel<<<(EE + 255) / 256, 256, 0, s>>>(eidx, EE, W.smax, W.score, W.denom);
  // scatter-add messages (neigh aliases y1; y1 dead after score pass)
  float* neigh = W.y1;
  fill_u32<<<((size_t)NN * DD + 255) / 256, 256, 0, s>>>((unsigned*)neigh, 0u, (size_t)NN * DD);
  edge_msg_kernel<<<(EE + 7) / 8, 256, 0, s>>>(eidx, etype, EE, W.z, W.relg, W.score,
                                               W.denom, W.deginv, neigh);
  // h = (neigh + self_msg)/2 ; BN stats ; BN + tanh
  fill_u32<<<1, 256, 0, s>>>((unsigned*)W.gsum, 0u, 256);  // gsum|gsumsq contiguous
  node_h_stats<<<(NN + 63) / 64, 128, 0, s>>>(neigh, W.z, W.relg + (size_t)RRR * DD,
                                              NN, W.gsum, W.gsumsq);
  bn_finalize<<<1, 128, 0, s>>>(W.gsum, W.gsumsq, gamma, beta, NN, W.bnsc, W.bnsh);
  bn_apply_tanh<<<((size_t)NN * DD + 255) / 256, 256, 0, s>>>(neigh, W.bnsc, W.bnsh, out,
                                                              (size_t)NN * DD);
}

extern "C" void kernel_launch(void* const* d_in, const int* in_sizes, int n_in,
                              void* d_out, int out_size, void* d_ws, size_t ws_size,
                              hipStream_t stream)
{
  (void)in_sizes; (void)n_in; (void)out_size; (void)ws_size;
  const float* ent_info  = (const float*)d_in[0];   // [N,300]
  const float* ent_comp  = (const float*)d_in[1];   // [N,128]
  const float* rel_comp  = (const float*)d_in[2];   // [500,128]
  const float* rel_align = (const float*)d_in[3];   // [500,128]
  const float* name_lin  = (const float*)d_in[4];   // [300,128]
  const float* U1        = (const float*)d_in[5];   // [256,128]
  const float* U2        = (const float*)d_in[6];   // [256,128]
  const float* L11       = (const float*)d_in[7];   // [128,128]
  const float* L12       = (const float*)d_in[8];   // [128,128]
  const float* AL        = (const float*)d_in[9];   // [384,128]
  const float* cw1       = (const float*)d_in[10];  // [3,128,128]
  const float* cw2       = (const float*)d_in[11];
  const float* cgcn      = (const float*)d_in[12];
  const float* cloop     = (const float*)d_in[13];  // [3,128]
  const float* cwatt     = (const float*)d_in[14];  // [3,256,128]
  const float* caatt     = (const float*)d_in[15];  // [3,128]
  const float* cgam      = (const float*)d_in[16];  // [3,128]
  const float* cbet      = (const float*)d_in[17];
  const int*   eidx      = (const int*)d_in[18];    // [2,E]: row0=dst, row1=src
  const int*   etype     = (const int*)d_in[19];    // [E]
  float*       out       = (float*)d_out;           // [N,128]

  // carve workspace
  char* p = (char*)d_ws;
  auto carve = [&](size_t bytes) -> void* {
    void* r = p; p += (bytes + 255) & ~(size_t)255; return r;
  };
  const size_t NB = (size_t)NN * DD * sizeof(float);
  const size_t RB = (size_t)(RRR + 1) * DD * sizeof(float);
  Ws W;
  W.B0       = (float*)carve(NB);
  W.B1       = (float*)carve(NB);
  W.B2       = (float*)carve(NB);
  W.B3       = (float*)carve(NB);
  W.y1       = (float*)carve(NB);
  W.y2       = (float*)carve(NB);
  W.z        = (float*)carve(NB);
  W.info_pad = (float*)carve((size_t)NN * NAMEDP * sizeof(float));
  W.bpack    = (unsigned short*)carve((size_t)NAMEDP * DD * sizeof(unsigned short));
  W.score    = (float*)carve((size_t)EE * sizeof(float));
  W.denom    = (float*)carve((size_t)NN * sizeof(float));
  W.deg      = (float*)carve((size_t)NN * sizeof(float));
  W.deginv   = (float*)carve((size_t)NN * sizeof(float));
  W.smax     = (unsigned*)carve((size_t)NN * sizeof(unsigned));
  W.relcat   = (float*)carve(RB);
  W.reltmp   = (float*)carve(RB);
  W.rel2     = (float*)carve(RB);
  W.relw     = (float*)carve(RB);
  W.relg     = (float*)carve(RB);
  W.arel_t   = (float*)carve(RB);
  W.arel     = (float*)carve(RB);
  W.gsum     = (float*)carve(256 * sizeof(float));   // gsum[128] | gsumsq[128]
  W.gsumsq   = W.gsum + 128;
  W.bnsc     = (float*)carve(DD * sizeof(float));
  W.bnsh     = (float*)carve(DD * sizeof(float));

  const int WW = DD * DD;  // 128*128

  // --- degrees (same dst for all convs; compute once)
  fill_u32<<<(NN + 255) / 256, 256, 0, stream>>>((unsigned*)W.deg, 0u, NN);
  deg_count<<<(EE + 255) / 256, 256, 0, stream>>>(eidx, EE, W.deg);
  deg_inv_kernel<<<(NN + 255) / 256, 256, 0, stream>>>(W.deg, W.deginv, NN);

  // --- init_align = [normalize(init_comp) | info@name_linear] @ U1
  pad_copy_kernel<<<((size_t)NN * NAMEDP + 255) / 256, 256, 0, stream>>>(
      ent_info, W.info_pad, NN, NAMED, NAMEDP);
  normalize_rows<<<(NN + 7) / 8, 256, 0, stream>>>(ent_comp, W.B3, NN);
  gemm(stream, W, W.info_pad, NN, NAMEDP, NAMED, name_lin, W.y1, 0);  // name_proj -> y1
  gemm(stream, W, W.B3, NN, DD, DD, U1, W.B0, 0);                     // top half
  gemm(stream, W, W.y1, NN, DD, DD, U1 + DD * DD, W.B0, 1);           // += bottom half

  // --- conv0 (alignment): x=init_align -> a1 (B1)
  run_conv(stream, W, W.B0, rel_align,
           cw1 + 0 * WW, cw2 + 0 * WW, cgcn + 0 * WW, cloop + 0 * DD,
           cwatt + 0 * 2 * WW, caatt + 0 * DD, cgam + 0 * DD, cbet + 0 * DD,
           eidx, etype, W.B1);

  // --- conv1 (completion): x=init_comp -> c1 (B2)
  run_conv(stream, W, ent_comp, rel_comp,
           cw1 + 1 * WW, cw2 + 1 * WW, cgcn + 1 * WW, cloop + 1 * DD,
           cwatt + 1 * 2 * WW, caatt + 1 * DD, cgam + 1 * DD, cbet + 1 * DD,
           eidx, etype, W.B2);

  // --- in2 = [normalize(c1) | a1] @ U2  (into B2; normalized c1 in B3)
  normalize_rows<<<(NN + 7) / 8, 256, 0, stream>>>(W.B2, W.B3, NN);
  gemm(stream, W, W.B3, NN, DD, DD, U2, W.B2, 0);
  gemm(stream, W, W.B1, NN, DD, DD, U2 + DD * DD, W.B2, 1);

  // --- align_rel = leaky(rel_align @ L11) @ L12
  gemm(stream, W, rel_align, RRR, DD, DD, L11, W.arel_t, 0, 1);
  gemm(stream, W, W.arel_t, RRR, DD, DD, L12, W.arel, 0);

  // --- conv2 (alignment): x=in2, rel=align_rel -> a2 (B3)
  run_conv(stream, W, W.B2, W.arel,
           cw1 + 2 * WW, cw2 + 2 * WW, cgcn + 2 * WW, cloop + 2 * DD,
           cwatt + 2 * 2 * WW, caatt + 2 * DD, cgam + 2 * DD, cbet + 2 * DD,
           eidx, etype, W.B3);

  // --- out = [init_align | a1 | a2] @ all_linear_completion
  gemm(stream, W, W.B0, NN, DD, DD, AL, out, 0);
  gemm(stream, W, W.B1, NN, DD, DD, AL + DD * DD, out, 1);
  gemm(stream, W, W.B3, NN, DD, DD, AL + 2 * DD * DD, out, 1);
}
